// TSPModel_30099130811107
// MI455X (gfx1250) — compile-verified, matrix operand order checked by reading, not run on hardware
//
#include <hip/hip_runtime.h>

// ---------------------------------------------------------------- types ----
typedef _Float16 v16h __attribute__((ext_vector_type(16)));
typedef _Float16 v8h  __attribute__((ext_vector_type(8)));
typedef float    v8f  __attribute__((ext_vector_type(8)));

#define WMMA_F16(a, b, c) \
    __builtin_amdgcn_wmma_f32_16x16x32_f16(false, (a), false, (b), (short)0, (c), false, false)
#define CAT16(lo, hi) \
    __builtin_shufflevector((lo), (hi), 0,1,2,3,4,5,6,7,8,9,10,11,12,13,14,15)

// ------------------------------------------------------------- problem -----
constexpr int Bb  = 32;
constexpr int Nn  = 1000;
constexpr int Ss  = 200;
constexpr int Dd  = 128;
constexpr int Hh  = 8;
constexpr int DK  = 16;
constexpr int FF  = 512;
constexpr int NL  = 3;
constexpr int HK  = 128;          // H*DK
constexpr int Lr  = Nn - Ss + 2;  // 802 real rows
constexpr int Lp  = 816;          // padded to 16
constexpr int Uu  = Nn - Ss;      // 800 unselected
constexpr int Mp  = Bb * Lp;      // 26112 rows, /16 = 1632 tiles

// ---------------------------------------------------------------------------
// 1) unselect[b][0..799] = sorted node ids not in sel[b]
// ---------------------------------------------------------------------------
__global__ void k_unselect(const int* __restrict__ sel, int* __restrict__ uns) {
    int b = blockIdx.x;
    __shared__ unsigned char flag[Nn];
    for (int n = threadIdx.x; n < Nn; n += blockDim.x) flag[n] = 0;
    __syncthreads();
    for (int s = threadIdx.x; s < Ss; s += blockDim.x) flag[sel[b * Ss + s]] = 1;
    __syncthreads();
    if (threadIdx.x == 0) {
        int c = 0;
        for (int n = 0; n < Nn; ++n)
            if (!flag[n]) uns[b * Uu + (c++)] = n;
    }
}

// ---------------------------------------------------------------------------
// 2) assemble x0 (B, Lp, D): row0 = first@Wfirst+bf, rows 1..800 = left,
//    row801 = last@Wlast+bl, rows 802..815 = 0.  Write f32 + f16 copies.
// ---------------------------------------------------------------------------
__global__ void k_assemble(const float* __restrict__ data, const int* __restrict__ sel,
                           const int* __restrict__ uns,
                           const float* __restrict__ Wf, const float* __restrict__ bf,
                           const float* __restrict__ Wl, const float* __restrict__ bl,
                           float* __restrict__ xf, _Float16* __restrict__ xh) {
    int b = blockIdx.x;
    int d = threadIdx.x;   // blockDim.x == 128
    int s0 = sel[b * Ss + 0];
    int sl = sel[b * Ss + Ss - 1];
    const float* rowF = data + ((size_t)b * Nn + s0) * Dd;
    const float* rowL = data + ((size_t)b * Nn + sl) * Dd;
    float af = bf[d], al = bl[d];
    for (int k = 0; k < Dd; ++k) {
        af += rowF[k] * Wf[k * Dd + d];
        al += rowL[k] * Wl[k * Dd + d];
    }
    size_t base = (size_t)b * Lp * Dd;
    xf[base + 0 * Dd + d] = af;            xh[base + 0 * Dd + d] = (_Float16)af;
    xf[base + (Lr - 1) * Dd + d] = al;     xh[base + (Lr - 1) * Dd + d] = (_Float16)al;
    for (int j = 0; j < Uu; ++j) {
        float v = data[((size_t)b * Nn + uns[b * Uu + j]) * Dd + d];
        xf[base + (1 + j) * Dd + d] = v;
        xh[base + (1 + j) * Dd + d] = (_Float16)v;
    }
    for (int i = Lr; i < Lp; ++i) {
        xf[base + (size_t)i * Dd + d] = 0.f;
        xh[base + (size_t)i * Dd + d] = (_Float16)0.f;
    }
}

// ---------------------------------------------------------------------------
// 3) pack f32 weight (K x N row-major) into WMMA B-operand lane order (f16):
//    P[((kt*NT+nt)*32+lane)*16+e] = W[kt*32 + Koff(e,lane)][nt*16 + lane%16]
// ---------------------------------------------------------------------------
__global__ void k_pack(const float* __restrict__ W, _Float16* __restrict__ P,
                       int K, int Ncols) {
    int idx = blockIdx.x * blockDim.x + threadIdx.x;
    if (idx >= K * Ncols) return;
    int e    = idx & 15;
    int lane = (idx >> 4) & 31;
    int tile = idx >> 9;
    int NT = Ncols >> 4;
    int kt = tile / NT, nt = tile % NT;
    int half = lane >> 4;
    int Koff = (e & 1) + (((e >> 1) & 3) << 1) + (half << 3) + ((e >> 3) << 4);
    P[idx] = (_Float16)W[(kt * 32 + Koff) * Ncols + nt * 16 + (lane & 15)];
}

// ---------------------------------------------------------------------------
// 4) generic WMMA GEMM: C(MxN) = A16(MxK) @ Bpacked(KxN) [+bias][+resid][relu]
//    one wave -> 16 rows x 128 cols (8 accumulators); 32-bit pointer stepping
// ---------------------------------------------------------------------------
template <int NTW>     // n-tiles per wave (8 -> 128 cols)
__global__ void k_gemm(const _Float16* __restrict__ A, const _Float16* __restrict__ Bp,
                       const float* __restrict__ bias, const float* __restrict__ resid,
                       float* __restrict__ Cf, _Float16* __restrict__ Ch,
                       int M, int K, int Ncols, int relu) {
    int wave = (blockIdx.x * blockDim.x + threadIdx.x) >> 5;
    int lane = threadIdx.x & 31;
    const int nchunks = Ncols / (16 * NTW);
    int mt = wave / nchunks;
    int nc = wave % nchunks;
    if (mt >= (M >> 4)) return;
    const int lm   = lane & 15;
    const int half = lane >> 4;
    const int nt0  = nc * NTW;
    const int NT   = Ncols >> 4;

    v8f acc[NTW] = {};
    const int KT = K >> 5;
    // 32-bit-friendly pointer stepping: all buffers < 2GB
    const _Float16* aptr = A + (unsigned)(mt * 16 + lm) * (unsigned)K + 8 * half;
    const _Float16* bptr = Bp + nt0 * 512 + lane * 16;   // tile stride = 512 halves
    const int bstep = NT * 512;                          // per-kt stride in halves
    for (int kt = 0; kt < KT; ++kt) {
        v8h alo = *(const v8h*)(aptr);
        v8h ahi = *(const v8h*)(aptr + 16);
        v16h a = CAT16(alo, ahi);
#pragma unroll
        for (int u = 0; u < NTW; ++u) {
            v16h b = *(const v16h*)(bptr + u * 512);     // folds to imm offset
            acc[u] = WMMA_F16(a, b, acc[u]);
        }
        aptr += 32;
        bptr += bstep;
    }
#pragma unroll
    for (int u = 0; u < NTW; ++u) {
        int ncol = (nt0 + u) * 16 + lm;
        float bv = bias ? bias[ncol] : 0.f;
#pragma unroll
        for (int r = 0; r < 8; ++r) {
            unsigned rrow = mt * 16 + r + 8 * half;
            unsigned idx = rrow * (unsigned)Ncols + ncol;
            float v = acc[u][r] + bv;
            if (resid) v += resid[idx];
            if (relu)  v = fmaxf(v, 0.f);
            if (Cf) Cf[idx] = v;
            if (Ch) Ch[idx] = (_Float16)v;
        }
    }
}

static void launch_gemm(const _Float16* A, const _Float16* Bp, const float* bias,
                        const float* resid, float* Cf, _Float16* Ch,
                        int M, int K, int Ncols, int relu, hipStream_t st) {
    int waves = (M / 16) * (Ncols / 128);
    int threads = waves * 32;
    int grid = (threads + 255) / 256;
    k_gemm<8><<<grid, 256, 0, st>>>(A, Bp, bias, resid, Cf, Ch, M, K, Ncols, relu);
}

// ---------------------------------------------------------------------------
// 5) flash attention: one wave per (b, h, 16-row q tile). DK=16 padded to K=32.
// ---------------------------------------------------------------------------
__global__ void k_attn(const _Float16* __restrict__ q, const _Float16* __restrict__ k,
                       const _Float16* __restrict__ v, _Float16* __restrict__ o) {
    int blk = blockIdx.x;
    const int IT = Lp / 16;               // 51
    int it = blk % IT; blk /= IT;
    int h  = blk % Hh;
    int b  = blk / Hh;
    int lane = threadIdx.x;
    int lm = lane & 15, half = lane >> 4;
    __shared__ __align__(16) _Float16 pLDS[16 * 16];

    const size_t base = (size_t)b * Lp * HK + h * DK;
    v8h z8 = {};
    // Q A-operand (constant over j loop): row it*16+lm, halves [8*half .. +7]
    const v8h* qp = (const v8h*)(q + base + (size_t)(it * 16 + lm) * HK + 8 * half);
    v16h aq = CAT16(qp[0], z8);

    float mo[8], lsum[8];
    v8f acc = {};
#pragma unroll
    for (int r = 0; r < 8; ++r) { mo[r] = -1e30f; lsum[r] = 0.f; }

    for (int jt = 0; jt < IT; ++jt) {
        // K^T B-operand: col n=lm is key row jt*16+lm, rows K = dk
        const v8h* kp = (const v8h*)(k + base + (size_t)(jt * 16 + lm) * HK + 8 * half);
        v16h bk = CAT16(kp[0], z8);
        v8f c0 = {};
        v8f s = WMMA_F16(aq, bk, c0);

        int j = jt * 16 + lm;
        float mx[8];
#pragma unroll
        for (int r = 0; r < 8; ++r) {
            float sv = s[r] * 0.25f;                       // 1/sqrt(DK)
            s[r] = (j < Lr) ? sv : -1e30f;
            mx[r] = s[r];
        }
#pragma unroll
        for (int msk = 1; msk < 16; msk <<= 1)
#pragma unroll
            for (int r = 0; r < 8; ++r)
                mx[r] = fmaxf(mx[r], __shfl_xor(mx[r], msk, 32));
        float al[8], ps[8];
#pragma unroll
        for (int r = 0; r < 8; ++r) {
            float mn = fmaxf(mo[r], mx[r]);
            al[r] = __expf(mo[r] - mn);
            s[r]  = __expf(s[r] - mn);
            mo[r] = mn;
            ps[r] = s[r];
        }
#pragma unroll
        for (int msk = 1; msk < 16; msk <<= 1)
#pragma unroll
            for (int r = 0; r < 8; ++r)
                ps[r] += __shfl_xor(ps[r], msk, 32);
#pragma unroll
        for (int r = 0; r < 8; ++r) {
            lsum[r] = lsum[r] * al[r] + ps[r];
            acc[r] *= al[r];
        }
        // transpose P through LDS: C-layout -> A-operand layout
        __syncthreads();
#pragma unroll
        for (int r = 0; r < 8; ++r)
            pLDS[(r + 8 * half) * 16 + lm] = (_Float16)s[r];
        __syncthreads();
        const v8h* pp = (const v8h*)(pLDS + lm * 16 + 8 * half);
        v16h pa = CAT16(pp[0], z8);
        // V B-operand: col n=lm (dk), rows K = j in tile
        v8h vlo;
#pragma unroll
        for (int e = 0; e < 8; ++e)
            vlo[e] = v[base + (size_t)(jt * 16 + 8 * half + e) * HK + lm];
        v16h bv = CAT16(vlo, z8);
        acc = WMMA_F16(pa, bv, acc);
    }
#pragma unroll
    for (int r = 0; r < 8; ++r) {
        int i = it * 16 + r + 8 * half;
        float val = (i < Lr) ? acc[r] / lsum[r] : 0.f;
        o[base + (size_t)i * HK + lm] = (_Float16)val;
    }
}

// ---------------------------------------------------------------------------
// 6) logits[b][i] = x[b,i,:] . Wfin + bfin   (one wave per row)
// ---------------------------------------------------------------------------
__global__ void k_logits(const float* __restrict__ xf, const float* __restrict__ Wfin,
                         const float* __restrict__ bfin, float* __restrict__ lg) {
    int wave = (blockIdx.x * blockDim.x + threadIdx.x) >> 5;
    int lane = threadIdx.x & 31;
    if (wave >= Bb * Lr) return;
    int b = wave / Lr, i = wave % Lr;
    const float* xr = xf + ((size_t)b * Lp + i) * Dd;
    float s = 0.f;
    for (int d = lane; d < Dd; d += 32) s += xr[d] * Wfin[d];
#pragma unroll
    for (int msk = 16; msk; msk >>= 1) s += __shfl_xor(s, msk, 32);
    if (lane == 0) lg[b * Lr + i] = s + bfin[0];
}

// ---------------------------------------------------------------------------
// 7) + attention bias, endpoint mask, softmax, prop fixup, scatter to output
// ---------------------------------------------------------------------------
__global__ void k_post(const float* __restrict__ ab, const int* __restrict__ sel,
                       const int* __restrict__ uns, float* __restrict__ lg,
                       float* __restrict__ out) {
    int b = blockIdx.x;
    int t = threadIdx.x;                 // 256 threads
    __shared__ float red[256];
    float* row = lg + b * Lr;
    int last = sel[b * Ss + Ss - 1];
    const float* abrow = ab + ((size_t)b * Nn + last) * Nn;
    for (int j = t; j < Uu; j += 256) row[1 + j] += abrow[uns[b * Uu + j]];
    __syncthreads();
    // softmax over interior (ends are -inf)
    float mx = -3.4e38f;
    for (int j = 1 + t; j <= Uu; j += 256) mx = fmaxf(mx, row[j]);
    red[t] = mx; __syncthreads();
    for (int s2 = 128; s2; s2 >>= 1) { if (t < s2) red[t] = fmaxf(red[t], red[t + s2]); __syncthreads(); }
    mx = red[0]; __syncthreads();
    float sm = 0.f;
    for (int j = 1 + t; j <= Uu; j += 256) sm += __expf(row[j] - mx);
    red[t] = sm; __syncthreads();
    for (int s2 = 128; s2; s2 >>= 1) { if (t < s2) red[t] += red[t + s2]; __syncthreads(); }
    sm = red[0]; __syncthreads();
    for (int n = t; n < Nn; n += 256) out[(size_t)b * Nn + n] = -2.0f;
    __syncthreads();
    for (int j = t; j < Uu; j += 256) {
        float p = __expf(row[1 + j] - mx) / sm;
        if (p <= 1e-5f) p += 1e-7f;
        out[(size_t)b * Nn + uns[b * Uu + j]] = p;
    }
}

// ---------------------------------------------------------------- launch ---
extern "C" void kernel_launch(void* const* d_in, const int* in_sizes, int n_in,
                              void* d_out, int out_size, void* d_ws, size_t ws_size,
                              hipStream_t stream) {
    const float* data   = (const float*)d_in[0];
    const float* abias  = (const float*)d_in[1];
    const float* Wfirst = (const float*)d_in[2];
    const float* bfirst = (const float*)d_in[3];
    const float* Wlast  = (const float*)d_in[4];
    const float* blast  = (const float*)d_in[5];
    const float* Wq     = (const float*)d_in[6];
    const float* Wk     = (const float*)d_in[7];
    const float* Wv     = (const float*)d_in[8];
    const float* Wo     = (const float*)d_in[9];
    const float* bo     = (const float*)d_in[10];
    const float* W1     = (const float*)d_in[11];
    const float* b1     = (const float*)d_in[12];
    const float* W2     = (const float*)d_in[13];
    const float* b2     = (const float*)d_in[14];
    const float* Wfin   = (const float*)d_in[15];
    const float* bfin   = (const float*)d_in[16];
    const int*   sel    = (const int*)d_in[17];
    float* out = (float*)d_out;

    // workspace carving (256B aligned)
    char* w = (char*)d_ws;
    size_t off = 0;
    auto alloc = [&](size_t bytes) -> void* {
        void* p = w + off;
        off = (off + bytes + 255) & ~(size_t)255;
        return p;
    };
    int*       uns  = (int*)alloc((size_t)Bb * Uu * 4);
    float*     xf   = (float*)alloc((size_t)Mp * Dd * 4);
    _Float16*  xh   = (_Float16*)alloc((size_t)Mp * Dd * 2);
    float*     o1f  = (float*)alloc((size_t)Mp * Dd * 4);
    _Float16*  o1h  = (_Float16*)alloc((size_t)Mp * Dd * 2);
    _Float16*  qh   = (_Float16*)alloc((size_t)Mp * HK * 2);
    _Float16*  kh   = (_Float16*)alloc((size_t)Mp * HK * 2);
    _Float16*  vh   = (_Float16*)alloc((size_t)Mp * HK * 2);
    _Float16*  oh   = (_Float16*)alloc((size_t)Mp * HK * 2);
    _Float16*  hh   = (_Float16*)alloc((size_t)Mp * FF * 2);
    _Float16*  pw   = (_Float16*)alloc((size_t)NL * (4 * 16384 + 2 * 65536) * 2);
    float*     lg   = (float*)alloc((size_t)Bb * Lr * 4);

    k_unselect<<<Bb, 256, 0, stream>>>(sel, uns);
    k_assemble<<<Bb, Dd, 0, stream>>>(data, sel, uns, Wfirst, bfirst, Wlast, blast, xf, xh);

    const size_t LSTRIDE = 4 * 16384 + 2 * 65536;   // halves per layer
    for (int l = 0; l < NL; ++l) {
        _Float16* pq = pw + l * LSTRIDE;
        k_pack<<<(Dd * HK + 255) / 256, 256, 0, stream>>>(Wq + (size_t)l * Dd * HK, pq,          Dd, HK);
        k_pack<<<(Dd * HK + 255) / 256, 256, 0, stream>>>(Wk + (size_t)l * Dd * HK, pq + 16384,  Dd, HK);
        k_pack<<<(Dd * HK + 255) / 256, 256, 0, stream>>>(Wv + (size_t)l * Dd * HK, pq + 32768,  Dd, HK);
        k_pack<<<(HK * Dd + 255) / 256, 256, 0, stream>>>(Wo + (size_t)l * HK * Dd, pq + 49152,  HK, Dd);
        k_pack<<<(Dd * FF + 255) / 256, 256, 0, stream>>>(W1 + (size_t)l * Dd * FF, pq + 65536,  Dd, FF);
        k_pack<<<(FF * Dd + 255) / 256, 256, 0, stream>>>(W2 + (size_t)l * FF * Dd, pq + 131072, FF, Dd);
    }

    for (int l = 0; l < NL; ++l) {
        const _Float16* pq = pw + l * LSTRIDE;
        const _Float16* pk = pq + 16384;
        const _Float16* pv = pq + 32768;
        const _Float16* po = pq + 49152;
        const _Float16* p1 = pq + 65536;
        const _Float16* p2 = pq + 131072;
        // QKV projections
        launch_gemm(xh, pq, nullptr, nullptr, nullptr, qh, Mp, Dd, HK, 0, stream);
        launch_gemm(xh, pk, nullptr, nullptr, nullptr, kh, Mp, Dd, HK, 0, stream);
        launch_gemm(xh, pv, nullptr, nullptr, nullptr, vh, Mp, Dd, HK, 0, stream);
        // attention
        k_attn<<<Bb * Hh * (Lp / 16), 32, 0, stream>>>(qh, kh, vh, oh);
        // out1 = x + o @ Wo + bo
        launch_gemm(oh, po, bo + (size_t)l * Dd, xf, o1f, o1h, Mp, HK, Dd, 0, stream);
        // h = relu(out1 @ W1 + b1)
        launch_gemm(o1h, p1, b1 + (size_t)l * FF, nullptr, nullptr, hh, Mp, Dd, FF, 1, stream);
        // x = out1 + h @ W2 + b2
        launch_gemm(hh, p2, b2 + (size_t)l * Dd, o1f, xf, xh, Mp, FF, Dd, 0, stream);
    }

    k_logits<<<(Bb * Lr * 32 + 255) / 256, 256, 0, stream>>>(xf, Wfin, bfin, lg);
    k_post<<<Bb, 256, 0, stream>>>(abias, sel, uns, lg, out);

    (void)in_sizes; (void)n_in; (void)out_size; (void)ws_size;
}
